// GQA_7868380086608
// MI455X (gfx1250) — matrix operand order repaired and checked by reading, not compiled, not run on hardware
//
#include <hip/hip_runtime.h>

// ---------------------------------------------------------------------------
// GQA attention block for MI455X (gfx1250), wave32 + WMMA bf16 (fp32 accum).
// ~120+ GFLOP vs ~0.3GB HBM traffic -> tensor-core bound. GEMMs use the CDNA5
// Tensor Data Mover (tensor_load_to_lds) with double-buffered LDS so the DMA
// of tile i+1 overlaps the WMMAs of tile i. Weights are pre-transposed once
// so both A and B tiles are contiguous [rows][K] 2D tiles (TDM-friendly,
// no per-block transpose traffic).
// Workspace requirement: ~130 MB.
// ---------------------------------------------------------------------------

#define Bn     2
#define Tn     2048
#define Cn     2048
#define NHn    16
#define NKVHn  4
#define HSn    128
#define QKVN   (Cn + 2 * NKVHn * HSn)   // 3072

typedef __attribute__((ext_vector_type(16))) __bf16 v16bf;
typedef __attribute__((ext_vector_type(8)))  __bf16 v8bf;
typedef __attribute__((ext_vector_type(8)))  float  v8f;
typedef __attribute__((ext_vector_type(4)))  unsigned int v4u;
typedef __attribute__((ext_vector_type(8)))  int v8i;
typedef __attribute__((ext_vector_type(4)))  int v4i;

__device__ __forceinline__ v16bf cat8(v8bf lo, v8bf hi) {
  return __builtin_shufflevector(lo, hi, 0,1,2,3,4,5,6,7,8,9,10,11,12,13,14,15);
}

// A-fragment (16x32, 16-bit), per ISA table:
// lane -> row m = lane%16 ; elems 0..7 : K=(lane/16)*8 + j ; elems 8..15 : K=16+(lane/16)*8 + j
__device__ __forceinline__ v16bf load_frag_a(const __bf16* base, int stride) {
  const int l = threadIdx.x & 31;
  const __bf16* p = base + (size_t)(l & 15) * stride + ((l >> 4) << 3);
  return cat8(*(const v8bf*)p, *(const v8bf*)(p + 16));
}

// B-fragment (32x16, 16-bit) read from an n-major buffer bt[n][k]
// (same K order as A-fragment):
// lane -> col n = lane%16 ; elems 0..7 : K=(lane/16)*8 + j ; elems 8..15 : K=16+(lane/16)*8 + j
__device__ __forceinline__ v16bf load_frag_b(const __bf16* base, int stride) {
  const int l = threadIdx.x & 31;
  const __bf16* p = base + (size_t)(l & 15) * stride + ((l >> 4) << 3);
  return cat8(*(const v8bf*)p, *(const v8bf*)(p + 16));
}

#define WMMA_BF16(a, b, c) \
  __builtin_amdgcn_wmma_f32_16x16x32_bf16(false, (a), false, (b), (short)0, (c), false, false)

#define RFL(x) ((unsigned int)__builtin_amdgcn_readfirstlane((int)(x)))

// ---------------------------------------------------------------------------
// TDM: issue a 2D tile load (tile_rows x tile_k bf16 elements, row stride =
// stride_elems) from global to LDS. D# built per CDNA5 ISA §8.3/§8.4:
//  group0: count=1 | lds_addr | global_addr[56:0] | type=2
//  group1: data_size=1(2B) | tensor_dim0/1 | tile_dim0/1 | tensor_dim0_stride
//  remaining groups: zero (2D tensor). Tracked with TENSORcnt.
// amdgpu-toolchain (clang-23) builtin arity: 6 args
//  (uint32x4 g0, int32x8 g1, int32x4, int32x4, int32x8, i32 cpol).
// ---------------------------------------------------------------------------
__device__ __forceinline__ void tdm_load_2d(unsigned int lds_off, unsigned long long gaddr,
                                            unsigned int tile_k, unsigned int tile_rows,
                                            unsigned int stride_elems, unsigned int dim_rows) {
  const unsigned int g0w0 = RFL(1u);                                   // count=1, user mode
  const unsigned int g0w1 = RFL(lds_off);                              // lds_addr (bytes)
  const unsigned int g0w2 = RFL((unsigned int)(gaddr & 0xFFFFFFFFu));  // global_addr[31:0]
  const unsigned int g0w3 = RFL(((unsigned int)(gaddr >> 32) & 0x1FFFFFFu) | (2u << 30)); // [56:32]|type=2
  const unsigned int d0 = stride_elems;   // tensor_dim0 >= tile_k (no OOB by construction)
  const unsigned int d1 = dim_rows;       // tensor_dim1 >= tile_rows
  const int g1w0 = (int)RFL(0x00010000u);                              // wg_mask=0, data_size=1 (2B)
  const int g1w1 = (int)RFL((d0 & 0xFFFFu) << 16);                     // tensor_dim0[15:0] @ [31:16]
  const int g1w2 = (int)RFL(((d0 >> 16) & 0xFFFFu) | ((d1 & 0xFFFFu) << 16));
  const int g1w3 = (int)RFL(((d1 >> 16) & 0xFFFFu) | (tile_k << 16));  // tile_dim0 @ [31:16]
  const int g1w4 = (int)RFL(tile_rows & 0xFFFFu);                      // tile_dim1; tile_dim2=0
  const int g1w5 = (int)RFL(stride_elems);                             // tensor_dim0_stride[31:0]
  const int g1w6 = 0;                                                  // stride[47:32]=0, dim1_stride lo=0
  const int g1w7 = 0;
  v4u g0 = { g0w0, g0w1, g0w2, g0w3 };
  v8i g1 = { g1w0, g1w1, g1w2, g1w3, g1w4, g1w5, g1w6, g1w7 };
  v4i gz4 = { 0, 0, 0, 0 };
  v8i gz8 = { 0, 0, 0, 0, 0, 0, 0, 0 };
  __builtin_amdgcn_tensor_load_to_lds(g0, g1, gz4, gz4, gz8, 0);
}

// ---------------------------------------------------------------------------
// fp32 -> bf16 conversion (plain)
// ---------------------------------------------------------------------------
__global__ __launch_bounds__(256) void cvt_f32_bf16(const float* __restrict__ in,
                                                    __bf16* __restrict__ out, int n) {
  int i = blockIdx.x * 256 + threadIdx.x;
  const int stride = gridDim.x * 256;
  for (; i < n; i += stride) out[i] = (__bf16)in[i];
}

// ---------------------------------------------------------------------------
// fp32 [K][N] -> bf16 transposed [N][K] (32x32 LDS tiles, done ONCE so the
// GEMM never re-transposes B).
// ---------------------------------------------------------------------------
__global__ __launch_bounds__(256) void cvt_transpose_bf16(const float* __restrict__ in,
                                                          __bf16* __restrict__ out,
                                                          int K, int N) {
  __shared__ float tile[32][33];
  const int kb = blockIdx.x * 32, nb = blockIdx.y * 32;
  const int tx = threadIdx.x & 31, ty = threadIdx.x >> 5;  // 32 x 8
  #pragma unroll
  for (int i = 0; i < 32; i += 8)
    tile[ty + i][tx] = in[(size_t)(kb + ty + i) * N + nb + tx];
  __syncthreads();
  #pragma unroll
  for (int i = 0; i < 32; i += 8)
    out[(size_t)(nb + ty + i) * K + kb + tx] = (__bf16)tile[tx][ty + i];
}

// ---------------------------------------------------------------------------
// TDM-fed WMMA GEMM: C[M,N] (fp32) = A[M,K] @ Bt[N,K]^T + bias[N], bf16 inputs.
// Block tile 128x128, 8 waves (4x2), K-step 32, double-buffered LDS.
// Wave 0 drives the Tensor Data Mover; DMA of step i+1 overlaps WMMA of step i.
// ---------------------------------------------------------------------------
__global__ __launch_bounds__(256) void gemm_bf16_wmma(const __bf16* __restrict__ A,
                                                      const __bf16* __restrict__ Bt,
                                                      const float* __restrict__ bias,
                                                      float* __restrict__ Co,
                                                      int M, int N, int K) {
  __shared__ __align__(16) __bf16 As[2][128][32];   // m-major, 2 x 8KB
  __shared__ __align__(16) __bf16 Bs[2][128][32];   // n-major, 2 x 8KB
  const int tid  = threadIdx.x;
  const int lane = tid & 31, wid = tid >> 5;
  const int wm = wid & 3, wn = wid >> 2;            // waves: 4 along M (32 rows), 2 along N (64 cols)
  const int mblk = blockIdx.x * 128, nblk = blockIdx.y * 128;
  const bool issuer = (wid == 0);                   // wave-uniform branch

  const unsigned int ldsA = (unsigned int)(uintptr_t)(&As[0][0][0]);
  const unsigned int ldsB = (unsigned int)(uintptr_t)(&Bs[0][0][0]);
  const unsigned long long gA = (unsigned long long)(uintptr_t)(A  + (size_t)mblk * K);
  const unsigned long long gB = (unsigned long long)(uintptr_t)(Bt + (size_t)nblk * K);

  if (issuer) {
    tdm_load_2d(ldsA, gA, 32, 128, (unsigned)K, (unsigned)M);
    tdm_load_2d(ldsB, gB, 32, 128, (unsigned)K, (unsigned)N);
  }

  v8f acc[2][4] = {};
  int buf = 0;
  for (int k0 = 0; k0 < K; k0 += 32, buf ^= 1) {
    if (issuer) __builtin_amdgcn_s_wait_tensorcnt(0);  // wave0: current buffer landed
    __syncthreads();                                   // publish to all waves; prev compute done
    if (issuer && (k0 + 32) < K) {                     // prefetch next tiles into other buffer
      const unsigned int boff = (unsigned)(buf ^ 1) * (128u * 32u * 2u);
      tdm_load_2d(ldsA + boff, gA + (unsigned long long)(k0 + 32) * 2u, 32, 128, (unsigned)K, (unsigned)M);
      tdm_load_2d(ldsB + boff, gB + (unsigned long long)(k0 + 32) * 2u, 32, 128, (unsigned)K, (unsigned)N);
    }

    v16bf af[2], bfr[4];
    #pragma unroll
    for (int mt = 0; mt < 2; ++mt) af[mt]  = load_frag_a(&As[buf][wm * 32 + mt * 16][0], 32);
    #pragma unroll
    for (int nt = 0; nt < 4; ++nt) bfr[nt] = load_frag_b(&Bs[buf][wn * 64 + nt * 16][0], 32);
    #pragma unroll
    for (int mt = 0; mt < 2; ++mt)
      #pragma unroll
      for (int nt = 0; nt < 4; ++nt)
        acc[mt][nt] = WMMA_BF16(af[mt], bfr[nt], acc[mt][nt]);
    __syncthreads();                                   // all reads of buf done before it is refilled
  }

  // C/D layout: element r -> M = r + 8*(lane/16), N = lane%16.
  const int rbase = (lane >> 4) << 3, n0 = lane & 15;
  #pragma unroll
  for (int mt = 0; mt < 2; ++mt)
    #pragma unroll
    for (int nt = 0; nt < 4; ++nt) {
      const int row = mblk + wm * 32 + mt * 16 + rbase;
      const int col = nblk + wn * 64 + nt * 16 + n0;
      float* dst = Co + (size_t)row * N + col;
      const float bv = bias[col];
      #pragma unroll
      for (int r = 0; r < 8; ++r) dst[(size_t)r * N] = acc[mt][nt][r] + bv;
    }
}

// ---------------------------------------------------------------------------
// RoPE + head split. qkv fp32 [B*T, 3072] ->
//   Qb bf16 [B][NH][T][HS], Kb bf16 [B][NKVH][T][HS], Vt bf16 [B][NKVH][HS][T]
// ---------------------------------------------------------------------------
__global__ __launch_bounds__(256) void rope_split(const float* __restrict__ qkv,
                                                  const float* __restrict__ fc,
                                                  const float* __restrict__ fs,
                                                  __bf16* __restrict__ Q,
                                                  __bf16* __restrict__ Kb,
                                                  __bf16* __restrict__ Vt) {
  const int bt = blockIdx.x;
  const int b = bt / Tn, t = bt % Tn;
  const float* row = qkv + (size_t)bt * QKVN;
  const int tid = threadIdx.x;
  const int HP = HSn / 2;  // 64 rotation pairs per head

  for (int p = tid; p < NHn * HP; p += 256) {
    int h = p / HP, i = p % HP;
    float re = row[h * HSn + 2 * i], im = row[h * HSn + 2 * i + 1];
    float c = fc[(size_t)t * HP + i], s = fs[(size_t)t * HP + i];
    size_t off = (((size_t)b * NHn + h) * Tn + t) * HSn + 2 * i;
    Q[off]     = (__bf16)(re * c - im * s);
    Q[off + 1] = (__bf16)(re * s + im * c);
  }
  for (int p = tid; p < NKVHn * HP; p += 256) {
    int g = p / HP, i = p % HP;
    float re = row[Cn + g * HSn + 2 * i], im = row[Cn + g * HSn + 2 * i + 1];
    float c = fc[(size_t)t * HP + i], s = fs[(size_t)t * HP + i];
    size_t off = (((size_t)b * NKVHn + g) * Tn + t) * HSn + 2 * i;
    Kb[off]     = (__bf16)(re * c - im * s);
    Kb[off + 1] = (__bf16)(re * s + im * c);
  }
  for (int e = tid; e < NKVHn * HSn; e += 256) {
    int g = e / HSn, d = e % HSn;
    float v = row[Cn + NKVHn * HSn + e];
    Vt[(((size_t)b * NKVHn + g) * HSn + d) * Tn + t] = (__bf16)v;
  }
}

// ---------------------------------------------------------------------------
// Flash attention (causal, GQA). grid = (B*NH, T/64), block = 128 (4 waves).
// Each wave: 16 query rows; 32-key steps; QK^T and PV through WMMA bf16;
// online softmax with half-wave (width-16) shuffles; P staged through
// wave-private LDS to convert C-layout -> A-layout. Output bf16 [B][T][C].
// K/V operands are L2-resident (512KB/head vs 192MB L2).
// ---------------------------------------------------------------------------
__global__ __launch_bounds__(128) void flash_attn_wmma(const __bf16* __restrict__ Q,
                                                       const __bf16* __restrict__ Kb,
                                                       const __bf16* __restrict__ Vt,
                                                       __bf16* __restrict__ Y) {
  const int bh = blockIdx.x;
  const int b = bh / NHn, h = bh % NHn, g = h / (NHn / NKVHn);
  const int wid = threadIdx.x >> 5, lane = threadIdx.x & 31;
  const int q0 = blockIdx.y * 64 + wid * 16;

  const __bf16* Qp = Q  + (((size_t)b * NHn   + h) * Tn + q0) * HSn;
  const __bf16* Kp = Kb + ((size_t)b * NKVHn + g) * Tn * HSn;   // [T][HS] (n-major for B-frags)
  const __bf16* Vp = Vt + ((size_t)b * NKVHn + g) * HSn * Tn;   // [HS][T] (n-major for B-frags)

  __shared__ __align__(16) __bf16 Plds[4][16][32];
  __bf16 (*P)[32] = Plds[wid];   // wave-private: no __syncthreads needed

  v16bf qf[4];
  #pragma unroll
  for (int c = 0; c < 4; ++c) qf[c] = load_frag_a(Qp + c * 32, HSn);

  v8f o[8] = {};
  float mr[8], lr[8];
  #pragma unroll
  for (int i = 0; i < 8; ++i) { mr[i] = -3.0e38f; lr[i] = 0.0f; }

  const float scale = 0.0883883476483184f;  // 1/sqrt(128)
  const int n0 = lane & 15, rb = (lane >> 4) << 3;
  const int send = q0 + 16;  // keys needed: 0 .. q0+15 (uniform per wave)

  for (int s0 = 0; s0 < send; s0 += 32) {
    v8f sa = {}, sb = {};
    #pragma unroll
    for (int c = 0; c < 4; ++c) {
      v16bf kb0 = load_frag_b(Kp + (size_t)s0 * HSn + c * 32, HSn);
      v16bf kb1 = load_frag_b(Kp + (size_t)(s0 + 16) * HSn + c * 32, HSn);
      sa = WMMA_BF16(qf[c], kb0, sa);
      sb = WMMA_BF16(qf[c], kb1, sb);
    }
    #pragma unroll
    for (int i = 0; i < 8; ++i) {
      const int q = q0 + rb + i;
      float v0 = sa[i] * scale; if (s0 + n0 > q)      v0 = -1.0e30f;
      float v1 = sb[i] * scale; if (s0 + 16 + n0 > q) v1 = -1.0e30f;
      float mx = fmaxf(v0, v1);
      #pragma unroll
      for (int off = 8; off >= 1; off >>= 1) mx = fmaxf(mx, __shfl_xor(mx, off, 16));
      const float mnew  = fmaxf(mr[i], mx);
      const float alpha = __expf(mr[i] - mnew);
      mr[i] = mnew;
      const float p0 = __expf(v0 - mnew);
      const float p1 = __expf(v1 - mnew);
      float rs = p0 + p1;
      #pragma unroll
      for (int off = 8; off >= 1; off >>= 1) rs += __shfl_xor(rs, off, 16);
      lr[i] = lr[i] * alpha + rs;
      #pragma unroll
      for (int nt = 0; nt < 8; ++nt) o[nt][i] *= alpha;
      P[rb + i][n0]      = (__bf16)p0;
      P[rb + i][16 + n0] = (__bf16)p1;
    }
    v16bf pf = load_frag_a(&P[0][0], 32);
    #pragma unroll
    for (int nt = 0; nt < 8; ++nt) {
      v16bf vf = load_frag_b(Vp + (size_t)(nt * 16) * Tn + s0, Tn);
      o[nt] = WMMA_BF16(pf, vf, o[nt]);
    }
  }

  #pragma unroll
  for (int i = 0; i < 8; ++i) {
    const int q = q0 + rb + i;
    const float inv = 1.0f / lr[i];
    #pragma unroll
    for (int nt = 0; nt < 8; ++nt) {
      const int d = nt * 16 + n0;
      Y[((size_t)b * Tn + q) * Cn + h * HSn + d] = (__bf16)(o[nt][i] * inv);
    }
  }
}

// ---------------------------------------------------------------------------
// Launch: x -> bf16 ; W -> bf16 transposed ; QKV GEMM ; RoPE/split ;
// flash attention ; proj GEMM.
// Inputs: x, freqs_cos, freqs_sin, W_attn, b_attn, W_proj, b_proj (all fp32).
// ---------------------------------------------------------------------------
extern "C" void kernel_launch(void* const* d_in, const int* in_sizes, int n_in,
                              void* d_out, int out_size, void* d_ws, size_t ws_size,
                              hipStream_t stream) {
  const float* x  = (const float*)d_in[0];
  const float* fc = (const float*)d_in[1];
  const float* fs = (const float*)d_in[2];
  const float* Wa = (const float*)d_in[3];
  const float* ba = (const float*)d_in[4];
  const float* Wp = (const float*)d_in[5];
  const float* bp = (const float*)d_in[6];
  float* out = (float*)d_out;

  // Workspace carve (~130 MB total), 256B aligned chunks.
  char* w = (char*)d_ws;
  auto carve = [&](size_t bytes) -> void* {
    char* p = w;
    w += (bytes + 255) & ~(size_t)255;
    return (void*)p;
  };
  const size_t M = (size_t)Bn * Tn;  // 4096
  __bf16* xb   = (__bf16*)carve(M * Cn * 2);
  __bf16* WabT = (__bf16*)carve((size_t)Cn * QKVN * 2);   // [QKVN][C]
  __bf16* WpbT = (__bf16*)carve((size_t)Cn * Cn * 2);     // [C][C] transposed
  float*  qkv  = (float*) carve(M * QKVN * 4);
  __bf16* Qb   = (__bf16*)carve((size_t)Bn * NHn   * Tn * HSn * 2);
  __bf16* Kbv  = (__bf16*)carve((size_t)Bn * NKVHn * Tn * HSn * 2);
  __bf16* Vt   = (__bf16*)carve((size_t)Bn * NKVHn * HSn * Tn * 2);
  __bf16* Yb   = (__bf16*)carve(M * Cn * 2);

  cvt_f32_bf16<<<2048, 256, 0, stream>>>(x, xb, (int)(M * Cn));
  cvt_transpose_bf16<<<dim3(Cn / 32, QKVN / 32), 256, 0, stream>>>(Wa, WabT, Cn, QKVN);
  cvt_transpose_bf16<<<dim3(Cn / 32, Cn / 32),   256, 0, stream>>>(Wp, WpbT, Cn, Cn);

  gemm_bf16_wmma<<<dim3((unsigned)(M / 128), QKVN / 128), 256, 0, stream>>>(
      xb, WabT, ba, qkv, (int)M, QKVN, Cn);

  rope_split<<<Bn * Tn, 256, 0, stream>>>(qkv, fc, fs, Qb, Kbv, Vt);

  flash_attn_wmma<<<dim3(Bn * NHn, Tn / 64), 128, 0, stream>>>(Qb, Kbv, Vt, Yb);

  gemm_bf16_wmma<<<dim3((unsigned)(M / 128), Cn / 128), 256, 0, stream>>>(
      Yb, WpbT, bp, out, (int)M, Cn, Cn);
}
